// CausalSelfAttention_11132555231381
// MI455X (gfx1250) — compile-verified, matrix operand order checked by reading, not run on hardware
//
#include <hip/hip_runtime.h>
#include <hip/hip_bf16.h>

// ---------------------------------------------------------------------------
// CDNA5 (gfx1250) causal self-attention:
//   - bf16 WMMA (V_WMMA_F32_16X16X32_BF16) everywhere, f32 accumulate
//   - flash-attention streaming softmax (no T*T materialization)
//   - Tensor Data Mover (TENSOR_LOAD_TO_LDS + s_wait_tensorcnt) for GEMM
//     operand tile staging, wave32
// ---------------------------------------------------------------------------

typedef __bf16 bf16_t;
typedef __attribute__((ext_vector_type(16))) __bf16 v16bf;
typedef __attribute__((ext_vector_type(8)))  float  v8f;
typedef __attribute__((ext_vector_type(4)))  unsigned int u32x4;
typedef __attribute__((ext_vector_type(8)))  int i32x8;
typedef __attribute__((ext_vector_type(4)))  int i32x4;

#if __has_builtin(__builtin_amdgcn_tensor_load_to_lds) && \
    __has_builtin(__builtin_amdgcn_s_wait_tensorcnt)
#define HAVE_TDM 1
#else
#define HAVE_TDM 0
#endif

__device__ __forceinline__ v8f wmma_bf16(v16bf a, v16bf b, v8f c) {
  // (neg_a, A, neg_b, B, c_mod, C, reuse_a, reuse_b)
  return __builtin_amdgcn_wmma_f32_16x16x32_bf16(false, a, false, b,
                                                 (short)0, c, false, false);
}

__device__ __forceinline__ v8f v8f_zero() {
  v8f v;
#pragma unroll
  for (int i = 0; i < 8; ++i) v[i] = 0.0f;
  return v;
}

constexpr int B_ = 4, T_ = 2048, C_ = 1024, H_ = 16, D_ = 64;
constexpr int BM = 64, BN = 64, BK = 32;

#if HAVE_TDM
// ---------------------------------------------------------------------------
// TDM: DMA one 2-D bf16 tile (rows x 32 elements) from global into LDS.
// LDS rows are padded by 1 DWORD per 16 DWORDs (pad_interval=3, pad_amount=0)
// to match the [BK+2] bf16 LDS row stride (68 bytes).
// D# layout per CDNA5 ISA.
// This toolchain exposes the 6-arg builtin:
//   (u32x4 g0, i32x8 g1, i32x4 g2, i32x4 g3, i32x8 g4, i32 cpol)
// ---------------------------------------------------------------------------
__device__ __forceinline__ void tdm_load_tile(void* lds, const bf16_t* gsrc,
                                              unsigned stride_elems,
                                              unsigned rows) {
  unsigned lds_off = (unsigned)(size_t)lds;          // low 32 bits = LDS offset
  unsigned long long ga = (unsigned long long)(size_t)gsrc;  // tile start

  u32x4 g0;
  g0[0] = 1u;                                   // count=1, user descriptor
  g0[1] = lds_off;                              // lds_addr  (bits 63:32)
  g0[2] = (unsigned)ga;                         // global_addr[31:0]
  g0[3] = (unsigned)(ga >> 32) | (2u << 30);    // global_addr[56:32] | type=2

  const unsigned dim1 = 1u << 20;               // generous tensor_dim1
  i32x8 g1;
  g1[0] = (int)((1u << 16)                      // data_size = 2 bytes
              | (1u << 20)                      // pad_enable
              | (3u << 22));                    // pad_interval: every 16 DWORDs
                                                // pad_amount=0 -> 1 DWORD pad
  g1[1] = (int)((stride_elems & 0xFFFFu) << 16);            // tensor_dim0 lo16
  g1[2] = (int)((stride_elems >> 16) | ((dim1 & 0xFFFFu) << 16));
  g1[3] = (int)((dim1 >> 16) | (32u << 16));    // tile_dim0 = 32 elements
  g1[4] = (int)(rows & 0xFFFFu);                // tile_dim1 = rows, tile_dim2=0
  g1[5] = (int)stride_elems;                    // tensor_dim0_stride lo32
  g1[6] = 0;
  g1[7] = 0;

  i32x4 z4;
  z4[0] = z4[1] = z4[2] = z4[3] = 0;            // groups 2/3 unused (2-D)
  i32x8 z8;
#pragma unroll
  for (int i = 0; i < 8; ++i) z8[i] = 0;

  __builtin_amdgcn_tensor_load_to_lds(g0, g1, z4, z4, z8, 0);
}
#endif

// ---------------------------------------------------------------------------
// fp32 -> bf16 bulk conversion (one-time; feeds TDM raw-byte tile copies)
// ---------------------------------------------------------------------------
__global__ void __launch_bounds__(256)
cvt_f32_bf16_kernel(const float* __restrict__ src, bf16_t* __restrict__ dst,
                    int n) {
  int i = blockIdx.x * 256 + threadIdx.x;
  if (i < n) dst[i] = (bf16_t)src[i];
}

// ---------------------------------------------------------------------------
// Tiled GEMM: Out[m,n] = sum_k X[m,k] * W[n,k] (+ bias[n]); X, W bf16.
// Block: 256 threads (8 waves). Operand tiles staged via TDM (wave 0 issues
// the two descriptors, s_wait_tensorcnt 0, then barrier). Each wave owns one
// 16-row M tile and two 16-col N tiles -> 2 WMMAs per k-step.
// ---------------------------------------------------------------------------
template <bool OUT_BF16>
__global__ void __launch_bounds__(256)
gemm_wmma_kernel(const bf16_t* __restrict__ X, const bf16_t* __restrict__ W,
                 const float* __restrict__ bias, void* __restrict__ Out,
                 int M, int N, int K)
{
  __shared__ bf16_t As[BM][BK + 2];
  __shared__ bf16_t Bs[BN][BK + 2];

  const int t    = threadIdx.x;
  const int lane = t & 31;
  const int wv   = t >> 5;
  const int hx   = lane >> 4;      // half-wave select (K packing)
  const int r    = lane & 15;
  const int mb   = (wv & 3) * 16;  // wave's M sub-tile
  const int nt   = (wv >> 2) * 32; // wave's N sub-tile pair
  const int row0 = blockIdx.y * BM;
  const int col0 = blockIdx.x * BN;

  v8f acc0 = v8f_zero();
  v8f acc1 = v8f_zero();

  for (int k0 = 0; k0 < K; k0 += BK) {
#if HAVE_TDM
    if (wv == 0) {
      tdm_load_tile(&As[0][0], X + (size_t)row0 * K + k0, (unsigned)K, BM);
      tdm_load_tile(&Bs[0][0], W + (size_t)col0 * K + k0, (unsigned)K, BN);
      __builtin_amdgcn_s_wait_tensorcnt(0);
    }
#else
#pragma unroll
    for (int j = 0; j < 8; ++j) {
      int li = t + 256 * j;
      int rr = li >> 5, cc = li & 31;
      As[rr][cc] = X[(size_t)(row0 + rr) * K + k0 + cc];
      Bs[rr][cc] = W[(size_t)(col0 + rr) * K + k0 + cc];
    }
#endif
    __syncthreads();

    // A fragment: lane r holds row (mb+r); K = hx*8..hx*8+7 and 16+hx*8..
    v16bf af, b0, b1;
#pragma unroll
    for (int j = 0; j < 8; ++j) {
      af[j]     = As[mb + r][hx * 8 + j];
      af[8 + j] = As[mb + r][16 + hx * 8 + j];
      b0[j]     = Bs[nt + r][hx * 8 + j];
      b0[8 + j] = Bs[nt + r][16 + hx * 8 + j];
      b1[j]     = Bs[nt + 16 + r][hx * 8 + j];
      b1[8 + j] = Bs[nt + 16 + r][16 + hx * 8 + j];
    }
    acc0 = wmma_bf16(af, b0, acc0);
    acc1 = wmma_bf16(af, b1, acc1);
    __syncthreads();
  }

  // C/D layout: VGPR i, lane -> row = i + 8*hx, col = r
#pragma unroll
  for (int i = 0; i < 8; ++i) {
    int grow = row0 + mb + i + 8 * hx;
    int gc0  = col0 + nt + r;
    int gc1  = col0 + nt + 16 + r;
    if constexpr (OUT_BF16) {
      bf16_t* o = (bf16_t*)Out;
      o[(size_t)grow * N + gc0] = (bf16_t)acc0[i];
      o[(size_t)grow * N + gc1] = (bf16_t)acc1[i];
    } else {
      float* o = (float*)Out;
      o[(size_t)grow * N + gc0] = acc0[i] + bias[gc0];
      o[(size_t)grow * N + gc1] = acc1[i] + bias[gc1];
    }
  }
}

// ---------------------------------------------------------------------------
// RoPE + layout split: qkv[B,T,3C](bf16) -> q,k [B,H,T,D], v transposed
// [B,H,D,T] (so P@V B-fragments are contiguous over keys).
// Reference: af[i] = 1024^(-i/15) for i<16, 0 for i in [16,32).
// ---------------------------------------------------------------------------
__global__ void __launch_bounds__(256)
rope_split_kernel(const bf16_t* __restrict__ qkv, bf16_t* __restrict__ q,
                  bf16_t* __restrict__ k, bf16_t* __restrict__ vt)
{
  int idx = blockIdx.x * 256 + threadIdx.x;   // B*T*H*32 threads
  int d2 = idx & 31;
  int hh = (idx >> 5) & (H_ - 1);
  int tt = (idx >> 9) & (T_ - 1);
  int b  = idx >> 20;

  float theta = 0.0f;
  if (d2 < 16) {
    float freq = __expf(-(float)d2 * (6.93147180559945f / 15.0f)); // ln(1024)/15
    theta = (float)tt * freq;
  }
  float c = __cosf(theta), s = __sinf(theta);

  size_t base = (size_t)(b * T_ + tt) * (3 * C_) + hh * D_;
  float q1 = (float)qkv[base + d2],          q2 = (float)qkv[base + d2 + 32];
  float k1 = (float)qkv[base + C_ + d2],     k2 = (float)qkv[base + C_ + d2 + 32];
  float v1 = (float)qkv[base + 2 * C_ + d2], v2 = (float)qkv[base + 2 * C_ + d2 + 32];

  size_t qkb = ((size_t)(b * H_ + hh) * T_ + tt) * D_;
  q[qkb + d2]      = (bf16_t)( q1 * c + q2 * s);
  q[qkb + d2 + 32] = (bf16_t)(-q1 * s + q2 * c);
  k[qkb + d2]      = (bf16_t)( k1 * c + k2 * s);
  k[qkb + d2 + 32] = (bf16_t)(-k1 * s + k2 * c);

  size_t vb = ((size_t)(b * H_ + hh) * D_ + d2) * T_ + tt;
  vt[vb]                   = (bf16_t)v1;
  vt[vb + (size_t)32 * T_] = (bf16_t)v2;
}

// ---------------------------------------------------------------------------
// Flash attention: grid (T/64, B*H), block 128 = 4 waves. Each wave owns 16
// query rows; streams 32 keys/iter: 4 S-WMMAs (D=64 reduction) + softmax
// update + 4 O-WMMAs. exp(S) relayout C->A via per-wave LDS tile (within-wave
// DS ops are in-order on CDNA5, no barrier needed).
// ---------------------------------------------------------------------------
__global__ void __launch_bounds__(128)
flash_attn_kernel(const bf16_t* __restrict__ q, const bf16_t* __restrict__ k,
                  const bf16_t* __restrict__ vt, bf16_t* __restrict__ y)
{
  __shared__ bf16_t Ps[4][16][36];

  const int lane = threadIdx.x & 31;
  const int wv   = threadIdx.x >> 5;
  const int hx   = lane >> 4;
  const int r    = lane & 15;
  const int bh   = blockIdx.y;
  const int b    = bh >> 4, head = bh & (H_ - 1);
  const int qb   = blockIdx.x * 64 + wv * 16;

  const bf16_t* Q = q  + (size_t)bh * T_ * D_;
  const bf16_t* K = k  + (size_t)bh * T_ * D_;
  const bf16_t* V = vt + (size_t)bh * D_ * T_;   // [D, T]

  // Q A-fragments for both 32-wide K-steps of the D=64 reduction
  v16bf qa[2];
#pragma unroll
  for (int kk = 0; kk < 2; ++kk) {
#pragma unroll
    for (int j = 0; j < 8; ++j) {
      qa[kk][j]     = Q[(size_t)(qb + r) * D_ + kk * 32 + hx * 8 + j];
      qa[kk][8 + j] = Q[(size_t)(qb + r) * D_ + kk * 32 + 16 + hx * 8 + j];
    }
  }

  float mrun[8], lrun[8];
  v8f o[4];
#pragma unroll
  for (int i = 0; i < 8; ++i) { mrun[i] = -3.0e38f; lrun[i] = 0.0f; }
#pragma unroll
  for (int dt = 0; dt < 4; ++dt) o[dt] = v8f_zero();

  const float scale = 0.125f;   // 1/sqrt(64)

  for (int kb = 0; kb < qb + 16; kb += 32) {
    v8f s0 = v8f_zero(), s1 = v8f_zero();
    v16bf kf;
#pragma unroll
    for (int kk = 0; kk < 2; ++kk) {
#pragma unroll
      for (int j = 0; j < 8; ++j) {       // keys kb..kb+15
        kf[j]     = K[(size_t)(kb + r) * D_ + kk * 32 + hx * 8 + j];
        kf[8 + j] = K[(size_t)(kb + r) * D_ + kk * 32 + 16 + hx * 8 + j];
      }
      s0 = wmma_bf16(qa[kk], kf, s0);
#pragma unroll
      for (int j = 0; j < 8; ++j) {       // keys kb+16..kb+31
        kf[j]     = K[(size_t)(kb + 16 + r) * D_ + kk * 32 + hx * 8 + j];
        kf[8 + j] = K[(size_t)(kb + 16 + r) * D_ + kk * 32 + 16 + hx * 8 + j];
      }
      s1 = wmma_bf16(qa[kk], kf, s1);
    }

    // Streaming softmax update. Row of element i is (qb + i + 8*hx); the 16
    // columns of that row live across one 16-lane half -> butterfly reduce.
#pragma unroll
    for (int i = 0; i < 8; ++i) {
      int grow = qb + i + 8 * hx;
      float f0 = s0[i] * scale;
      float f1 = s1[i] * scale;
      if (kb + r > grow)      f0 = -3.0e38f;   // causal mask
      if (kb + 16 + r > grow) f1 = -3.0e38f;
      float mx = fmaxf(f0, f1);
#pragma unroll
      for (int off = 1; off < 16; off <<= 1)
        mx = fmaxf(mx, __shfl_xor(mx, off, 32));
      float newm  = fmaxf(mrun[i], mx);
      float alpha = __expf(mrun[i] - newm);
      float p0 = __expf(f0 - newm);
      float p1 = __expf(f1 - newm);
      float rs = p0 + p1;
#pragma unroll
      for (int off = 1; off < 16; off <<= 1)
        rs += __shfl_xor(rs, off, 32);
      mrun[i] = newm;
      lrun[i] = lrun[i] * alpha + rs;
#pragma unroll
      for (int dt = 0; dt < 4; ++dt)
        o[dt][i] *= alpha;                      // rescale this row only
      Ps[wv][i + 8 * hx][r]      = (bf16_t)p0;  // C-layout -> LDS
      Ps[wv][i + 8 * hx][16 + r] = (bf16_t)p1;
    }

    // Read P back as an A-fragment (16 q-rows x 32 keys)
    v16bf pa;
#pragma unroll
    for (int j = 0; j < 8; ++j) {
      pa[j]     = Ps[wv][r][hx * 8 + j];
      pa[8 + j] = Ps[wv][r][16 + hx * 8 + j];
    }

    // O += P @ V ; B-fragment column n = V column (dt*16+r) = Vt row, keys
    // contiguous along T.
#pragma unroll
    for (int dt = 0; dt < 4; ++dt) {
      v16bf vb;
#pragma unroll
      for (int j = 0; j < 8; ++j) {
        vb[j]     = V[(size_t)(dt * 16 + r) * T_ + kb + hx * 8 + j];
        vb[8 + j] = V[(size_t)(dt * 16 + r) * T_ + kb + 16 + hx * 8 + j];
      }
      o[dt] = wmma_bf16(pa, vb, o[dt]);
    }
  }

  // Normalize and store y as [B, T, C] bf16 for the output projection.
#pragma unroll
  for (int i = 0; i < 8; ++i) {
    int grow  = qb + i + 8 * hx;
    float inv = 1.0f / lrun[i];
#pragma unroll
    for (int dt = 0; dt < 4; ++dt) {
      y[(size_t)(b * T_ + grow) * C_ + head * D_ + dt * 16 + r] =
          (bf16_t)(o[dt][i] * inv);
    }
  }
}

// ---------------------------------------------------------------------------
extern "C" void kernel_launch(void* const* d_in, const int* in_sizes, int n_in,
                              void* d_out, int out_size, void* d_ws, size_t ws_size,
                              hipStream_t stream)
{
  const float* x        = (const float*)d_in[0];
  const float* qkv_w    = (const float*)d_in[1];   // (3C, C) row-major
  const float* c_proj_w = (const float*)d_in[2];   // (C, C)
  const float* c_proj_b = (const float*)d_in[3];

  const int M     = B_ * T_;    // 8192
  const int N_qkv = 3 * C_;     // 3072
  const size_t HD = (size_t)B_ * H_ * T_ * D_;     // 8,388,608 elems

  bf16_t* xbf   = (bf16_t*)d_ws;
  bf16_t* wqkv  = xbf + (size_t)M * C_;
  bf16_t* wproj = wqkv + (size_t)N_qkv * C_;
  bf16_t* qkv   = wproj + (size_t)C_ * C_;
  bf16_t* q     = qkv + (size_t)M * N_qkv;
  bf16_t* k     = q + HD;
  bf16_t* vt    = k + HD;
  bf16_t* y     = vt + HD;

  dim3 blk256(256);

  // 0) One-time fp32 -> bf16 conversions (activations + weights) so GEMM
  //    operand tiles are raw-byte TDM copies.
  int n_x = M * C_, n_wq = N_qkv * C_, n_wp = C_ * C_;
  cvt_f32_bf16_kernel<<<(n_x  + 255) / 256, blk256, 0, stream>>>(x, xbf, n_x);
  cvt_f32_bf16_kernel<<<(n_wq + 255) / 256, blk256, 0, stream>>>(qkv_w, wqkv, n_wq);
  cvt_f32_bf16_kernel<<<(n_wp + 255) / 256, blk256, 0, stream>>>(c_proj_w, wproj, n_wp);

  // 1) QKV projection GEMM (bf16 out), TDM-staged tiles
  dim3 g1(N_qkv / BN, M / BM);  // 48 x 128
  gemm_wmma_kernel<true><<<g1, blk256, 0, stream>>>(
      xbf, wqkv, nullptr, qkv, M, N_qkv, C_);

  // 2) RoPE + split/transpose
  int rope_threads = B_ * T_ * H_ * 32;             // 2^22
  rope_split_kernel<<<rope_threads / 256, blk256, 0, stream>>>(qkv, q, k, vt);

  // 3) Flash attention
  dim3 ga(T_ / 64, B_ * H_);   // 32 x 64
  flash_attn_kernel<<<ga, dim3(128), 0, stream>>>(q, k, vt, y);

  // 4) Output projection (+bias), fp32 out, TDM-staged tiles
  dim3 g2(C_ / BN, M / BM);    // 16 x 128
  gemm_wmma_kernel<false><<<g2, blk256, 0, stream>>>(
      y, wproj, c_proj_b, d_out, M, C_, C_);
}
